// AttentionBlock_15573551416041
// MI455X (gfx1250) — matrix operand order repaired and checked by reading, not compiled, not run on hardware
//
#include <hip/hip_runtime.h>
#include <hip/hip_bf16.h>

typedef _Float16 half_t;
typedef __attribute__((ext_vector_type(16))) _Float16 v16h;
typedef __attribute__((ext_vector_type(8)))  _Float16 v8h;
typedef __attribute__((ext_vector_type(8)))  float    v8f;

#define BB 8
#define CC 256
#define NN 4096
#define GG 8

// ---------------------------------------------------------------------------
// WMMA helpers (CDNA5 16x16x32 f16 -> f32)
// ---------------------------------------------------------------------------
__device__ __forceinline__ v8f wmma_f16(v16h a, v16h b, v8f c) {
  return __builtin_amdgcn_wmma_f32_16x16x32_f16(false, a, false, b,
                                                (short)0, c, false, false);
}

// A fragment: 16x32 (MxK). Source row-major: element(m,k) at base[m*stride+k].
__device__ __forceinline__ v16h frag_a(const half_t* base, int stride) {
  const int lane = threadIdx.x & 31;
  const int r    = lane & 15;
  const int h8   = (lane >> 4) << 3;
  const half_t* p = base + (size_t)r * stride + h8;
  v8h lo = *(const v8h*)(p);
  v8h hi = *(const v8h*)(p + 16);
  return __builtin_shufflevector(lo, hi, 0,1,2,3,4,5,6,7,8,9,10,11,12,13,14,15);
}

// B fragment: 32x16 (KxN). Source stored TRANSPOSED: element(k,col) at
// base[col*stride + k]. lane L: col = L%16, K = 8*(L/16) + {0..7}, 16 + 8*(L/16) + {0..7}.
__device__ __forceinline__ v16h frag_b(const half_t* base, int stride) {
  const int lane = threadIdx.x & 31;
  const int c    = lane & 15;
  const int kb   = (lane >> 4) << 3;
  const half_t* p = base + (size_t)c * stride + kb;
  v8h lo = *(const v8h*)(p);
  v8h hi = *(const v8h*)(p + 16);
  return __builtin_shufflevector(lo, hi, 0,1,2,3,4,5,6,7,8,9,10,11,12,13,14,15);
}

// Async Global -> LDS copy of 16 bytes (CDNA5, tracked by ASYNCcnt; no VGPR
// round-trip). LDS address = low 32 bits of the generic shared pointer.
__device__ __forceinline__ void async_cp16(half_t* dst_lds, const half_t* src) {
  asm volatile("global_load_async_to_lds_b128 %0, %1, off"
               :
               : "v"((unsigned int)(unsigned long long)dst_lds),
                 "v"((unsigned long long)src)
               : "memory");
}
__device__ __forceinline__ void wait_async0() {
  asm volatile("s_wait_asynccnt 0x0" ::: "memory");
}

// ---------------------------------------------------------------------------
// K0: convert weights fp32 -> f16, transposed to [nout][k] (k contiguous)
// ---------------------------------------------------------------------------
__global__ void conv_weights(const float* __restrict__ wqkv,
                             const float* __restrict__ wproj,
                             half_t* __restrict__ wqkvt,
                             half_t* __restrict__ wpt) {
  int tid = blockIdx.x * 256 + threadIdx.x;
  if (tid < 768 * 256) {
    int nout = tid >> 8, kin = tid & 255;
    wqkvt[tid] = (half_t)wqkv[(size_t)kin * 768 + nout];
  }
  int t2 = tid - 768 * 256;
  if (t2 >= 0 && t2 < 256 * 256) {
    int nout = t2 >> 8, kin = t2 & 255;
    wpt[t2] = (half_t)wproj[(size_t)kin * 256 + nout];
  }
}

// ---------------------------------------------------------------------------
// K1: GroupNorm stats, one block per (b,g)
// ---------------------------------------------------------------------------
__global__ __launch_bounds__(256) void gn_stats(const float* __restrict__ x,
                                                float* __restrict__ mean,
                                                float* __restrict__ rstd) {
  const int bg = blockIdx.x;                       // b*8 + g
  const float* p = x + (size_t)bg * (32 * NN);     // 32 channels * 4096
  float s1 = 0.f, s2 = 0.f;
  for (int i = threadIdx.x; i < 32 * NN; i += 256) {
    float v = p[i];
    s1 += v; s2 += v * v;
  }
  __shared__ float a1[256], a2[256];
  a1[threadIdx.x] = s1; a2[threadIdx.x] = s2;
  __syncthreads();
  for (int off = 128; off > 0; off >>= 1) {
    if (threadIdx.x < off) {
      a1[threadIdx.x] += a1[threadIdx.x + off];
      a2[threadIdx.x] += a2[threadIdx.x + off];
    }
    __syncthreads();
  }
  if (threadIdx.x == 0) {
    const float cnt = 32.f * NN;
    float mu  = a1[0] / cnt;
    float var = a2[0] / cnt - mu * mu;
    mean[bg] = mu;
    rstd[bg] = rsqrtf(var + 1e-5f);
  }
}

// ---------------------------------------------------------------------------
// K2: apply GN + transpose to token-major f16: ht[b][n][c]
// ---------------------------------------------------------------------------
__global__ __launch_bounds__(256) void gn_apply(const float* __restrict__ x,
                                                const float* __restrict__ gamma,
                                                const float* __restrict__ beta,
                                                const float* __restrict__ mean,
                                                const float* __restrict__ rstd,
                                                half_t* __restrict__ ht) {
  const int b = blockIdx.y;
  const int n = blockIdx.x * 256 + threadIdx.x;
  const float* xb = x + (size_t)b * CC * NN;
  v8h buf[32];
#pragma unroll
  for (int cc = 0; cc < 32; ++cc) {
#pragma unroll
    for (int i = 0; i < 8; ++i) {
      const int c = cc * 8 + i;
      const float mu = mean[b * GG + (c >> 5)];
      const float rs = rstd[b * GG + (c >> 5)];
      float v = (xb[(size_t)c * NN + n] - mu) * rs * gamma[c] + beta[c];
      buf[cc][i] = (half_t)v;
    }
  }
  half_t* o = ht + ((size_t)b * NN + n) * CC;
#pragma unroll
  for (int cc = 0; cc < 32; ++cc) *(v8h*)(o + cc * 8) = buf[cc];
}

// ---------------------------------------------------------------------------
// K3: QKV GEMM (32768 x 256 x 768). Q scaled by 1/16.
//     q,k -> [b][n][d] f16 ; v -> transposed vt[b][d][n] f16
// ---------------------------------------------------------------------------
__global__ __launch_bounds__(256) void qkv_gemm(const half_t* __restrict__ ht,
                                                const half_t* __restrict__ wt,
                                                const float* __restrict__ bqkv,
                                                half_t* __restrict__ qo,
                                                half_t* __restrict__ ko,
                                                half_t* __restrict__ vto) {
  const int w    = threadIdx.x >> 5;
  const int lane = threadIdx.x & 31;
  const int row0 = blockIdx.y * 128 + w * 16;      // flattened b*N+n row
  const int col0 = blockIdx.x * 64;
  v8f acc[4] = {};
#pragma unroll
  for (int kk = 0; kk < 8; ++kk) {
    v16h a = frag_a(ht + (size_t)row0 * 256 + kk * 32, 256);
#pragma unroll
    for (int t = 0; t < 4; ++t) {
      v16h bf = frag_b(wt + (size_t)(col0 + t * 16) * 256 + kk * 32, 256);
      acc[t] = wmma_f16(a, bf, acc[t]);
    }
  }
  const int hi = lane >> 4, cl = lane & 15;
#pragma unroll
  for (int t = 0; t < 4; ++t) {
    const int nout = col0 + t * 16 + cl;
    const float bias = bqkv[nout];
#pragma unroll
    for (int r = 0; r < 8; ++r) {
      const int row = row0 + r + 8 * hi;
      const float v = acc[t][r] + bias;
      const int b = row >> 12, n = row & (NN - 1);
      if (nout < 256)      qo[(size_t)row * 256 + nout]            = (half_t)(v * 0.0625f);
      else if (nout < 512) ko[(size_t)row * 256 + (nout - 256)]    = (half_t)v;
      else                 vto[((size_t)b * 256 + (nout - 512)) * NN + n] = (half_t)v;
    }
  }
}

// ---------------------------------------------------------------------------
// K4: Flash attention. 8 waves/block; wave owns 16 query rows.
//     K/V tiles double-buffered in LDS via async Global->LDS copies.
// ---------------------------------------------------------------------------
__global__ __launch_bounds__(256) void flash_attn(const half_t* __restrict__ qg,
                                                  const half_t* __restrict__ kg,
                                                  const half_t* __restrict__ vtg,
                                                  half_t* __restrict__ ao) {
  __shared__ half_t kt[2][32 * 256];     // [key][d]  (B-transposed for QK^T)
  __shared__ half_t vs[2][256 * 32];     // [d][key]  (B-transposed for P*V)
  __shared__ half_t pst[8][16 * 32];     // per-wave P staging (C->A relayout)

  const int b    = blockIdx.y;
  const int tid  = threadIdx.x;
  const int w    = tid >> 5;
  const int lane = tid & 31;
  const int hi   = lane >> 4, cl = lane & 15;
  const int qrow0 = blockIdx.x * 128 + w * 16;

  // cooperative async tile fetch: this thread's slice of K and V tiles
  const int jrow = tid >> 3, part = tid & 7;
  const half_t* ksrc0 = kg + ((size_t)b * NN + jrow) * 256 + part * 32;
  const half_t* vsrc0 = vtg + ((size_t)b * 256 + tid) * NN;
  auto issue_tiles = [&](int j0, int buf) {
    const half_t* ks = ksrc0 + (size_t)j0 * 256;
    half_t* kd = &kt[buf][jrow * 256 + part * 32];
    async_cp16(kd,      ks);
    async_cp16(kd + 8,  ks + 8);
    async_cp16(kd + 16, ks + 16);
    async_cp16(kd + 24, ks + 24);
    const half_t* vsrc = vsrc0 + j0;
    half_t* vd = &vs[buf][tid * 32];
    async_cp16(vd,      vsrc);
    async_cp16(vd + 8,  vsrc + 8);
    async_cp16(vd + 16, vsrc + 16);
    async_cp16(vd + 24, vsrc + 24);
  };

  v16h qa[8];
#pragma unroll
  for (int kk = 0; kk < 8; ++kk)
    qa[kk] = frag_a(qg + ((size_t)b * NN + qrow0) * 256 + kk * 32, 256);

  v8f o[16] = {};
  float m[8], l[8];
#pragma unroll
  for (int r = 0; r < 8; ++r) { m[r] = -1e30f; l[r] = 0.f; }

  issue_tiles(0, 0);
  wait_async0();
  __syncthreads();

  for (int j0 = 0; j0 < NN; j0 += 32) {
    const int cur = (j0 >> 5) & 1;
    if (j0 + 32 < NN) issue_tiles(j0 + 32, cur ^ 1);   // prefetch next tile

    // S = Q * K^T  (two 16x16 column tiles of the 16x32 score block)
    v8f s0 = {}, s1 = {};
#pragma unroll
    for (int kk = 0; kk < 8; ++kk) {
      v16h b0 = frag_b(&kt[cur][kk * 32], 256);
      v16h b1 = frag_b(&kt[cur][16 * 256 + kk * 32], 256);
      s0 = wmma_f16(qa[kk], b0, s0);
      s1 = wmma_f16(qa[kk], b1, s1);
    }

    // online softmax; rows r+8*hi live in the same lane across all accum tiles
    float alpha[8];
#pragma unroll
    for (int r = 0; r < 8; ++r) {
      float mx = fmaxf(s0[r], s1[r]);
      mx = fmaxf(mx, __shfl_xor(mx, 1));
      mx = fmaxf(mx, __shfl_xor(mx, 2));
      mx = fmaxf(mx, __shfl_xor(mx, 4));
      mx = fmaxf(mx, __shfl_xor(mx, 8));
      const float mn = fmaxf(m[r], mx);
      const float p0 = __expf(s0[r] - mn);
      const float p1 = __expf(s1[r] - mn);
      float sum = p0 + p1;
      sum += __shfl_xor(sum, 1);
      sum += __shfl_xor(sum, 2);
      sum += __shfl_xor(sum, 4);
      sum += __shfl_xor(sum, 8);
      alpha[r] = __expf(m[r] - mn);
      l[r] = l[r] * alpha[r] + sum;
      m[r] = mn;
      // stage probabilities (C layout -> memory) for A-fragment reload
      pst[w][(r + 8 * hi) * 32 + cl]      = (half_t)p0;
      pst[w][(r + 8 * hi) * 32 + 16 + cl] = (half_t)p1;
    }
#pragma unroll
    for (int t = 0; t < 16; ++t)
#pragma unroll
      for (int r = 0; r < 8; ++r) o[t][r] *= alpha[r];

    asm volatile("s_wait_dscnt 0" ::: "memory");   // wave-local LDS RAW fence
    v16h pa = frag_a(&pst[w][0], 32);

    // O += P * V  over all 16 d-column tiles
#pragma unroll
    for (int t = 0; t < 16; ++t) {
      v16h bv = frag_b(&vs[cur][t * 16 * 32], 32);
      o[t] = wmma_f16(pa, bv, o[t]);
    }

    wait_async0();        // own async copies for next tile landed
    __syncthreads();      // whole block done reading cur / writing cur^1
  }

  float inv[8];
#pragma unroll
  for (int r = 0; r < 8; ++r) inv[r] = 1.f / l[r];
  half_t* op = ao + ((size_t)b * NN + qrow0) * 256;
#pragma unroll
  for (int t = 0; t < 16; ++t)
#pragma unroll
    for (int r = 0; r < 8; ++r)
      op[(size_t)(r + 8 * hi) * 256 + t * 16 + cl] = (half_t)(o[t][r] * inv[r]);
}

// ---------------------------------------------------------------------------
// K5: output projection + bias + residual -> fp32 [B,C,N]
// ---------------------------------------------------------------------------
__global__ __launch_bounds__(256) void proj_out(const half_t* __restrict__ ao,
                                                const half_t* __restrict__ wpt,
                                                const float* __restrict__ bproj,
                                                const float* __restrict__ x,
                                                float* __restrict__ out) {
  const int b    = blockIdx.y;
  const int w    = threadIdx.x >> 5;
  const int lane = threadIdx.x & 31;
  const int n0   = blockIdx.x * 128 + w * 16;
  v8f acc[16] = {};
#pragma unroll
  for (int kk = 0; kk < 8; ++kk) {
    v16h a = frag_a(ao + ((size_t)b * NN + n0) * 256 + kk * 32, 256);
#pragma unroll
    for (int t = 0; t < 16; ++t) {
      v16h bf = frag_b(wpt + (size_t)(t * 16) * 256 + kk * 32, 256);
      acc[t] = wmma_f16(a, bf, acc[t]);
    }
  }
  const int hi = lane >> 4, cl = lane & 15;
#pragma unroll
  for (int t = 0; t < 16; ++t) {
    const int c = t * 16 + cl;
    const float bias = bproj[c];
#pragma unroll
    for (int r = 0; r < 8; ++r) {
      const int n = n0 + r + 8 * hi;
      const size_t idx = ((size_t)b * CC + c) * NN + n;
      out[idx] = x[idx] + acc[t][r] + bias;
    }
  }
}

// ---------------------------------------------------------------------------
extern "C" void kernel_launch(void* const* d_in, const int* in_sizes, int n_in,
                              void* d_out, int out_size, void* d_ws, size_t ws_size,
                              hipStream_t stream) {
  const float* x      = (const float*)d_in[0];
  const float* gamma  = (const float*)d_in[1];
  const float* beta   = (const float*)d_in[2];
  const float* w_qkv  = (const float*)d_in[3];
  const float* b_qkv  = (const float*)d_in[4];
  const float* w_proj = (const float*)d_in[5];
  const float* b_proj = (const float*)d_in[6];
  float* out = (float*)d_out;

  char* ws = (char*)d_ws;
  float* mean = (float*)ws;               // 64 floats
  float* rstd = mean + 64;                // 64 floats
  const size_t TOK = (size_t)BB * NN * CC;   // 8,388,608 elements
  half_t* ht    = (half_t*)(ws + 1024);
  half_t* qbuf  = ht + TOK;
  half_t* kbuf  = qbuf + TOK;
  half_t* vtbuf = kbuf + TOK;
  half_t* wqkvt = vtbuf + TOK;            // 768*256
  half_t* wpt   = wqkvt + 768 * 256;      // 256*256
  half_t* ao    = ht;                     // ht is dead after qkv_gemm -> reuse

  conv_weights<<<dim3(1024), dim3(256), 0, stream>>>(w_qkv, w_proj, wqkvt, wpt);
  gn_stats<<<dim3(BB * GG), dim3(256), 0, stream>>>(x, mean, rstd);
  gn_apply<<<dim3(NN / 256, BB), dim3(256), 0, stream>>>(x, gamma, beta, mean, rstd, ht);
  qkv_gemm<<<dim3(768 / 64, (BB * NN) / 128), dim3(256), 0, stream>>>(ht, wqkvt, b_qkv,
                                                                      qbuf, kbuf, vtbuf);
  flash_attn<<<dim3(NN / 128, BB), dim3(256), 0, stream>>>(qbuf, kbuf, vtbuf, ao);
  proj_out<<<dim3(NN / 128, BB), dim3(256), 0, stream>>>(ao, wpt, b_proj, x, out);
}
